// T_67336497267186
// MI455X (gfx1250) — matrix operand order repaired and checked by reading, not compiled, not run on hardware
//
#include <hip/hip_runtime.h>

// out = x + exp(0.5*(relu(x@W1+b1)@W2+b2)) * eps
// B=131072, S=256, H=64. Memory-bound (~400MB @ 23.3TB/s ~= 17us); bf16 WMMA
// for both GEMMs, fp32 accumulate, single pass over x/eps/out.

#define SDIM 256
#define HDIM 64

typedef __attribute__((ext_vector_type(16))) __bf16         v16bf;
typedef __attribute__((ext_vector_type(8)))  float          v8f;
typedef __attribute__((ext_vector_type(4)))  float          v4f;
typedef __attribute__((ext_vector_type(4)))  int            v4i;
typedef __attribute__((ext_vector_type(8)))  unsigned short v8u;
typedef __attribute__((ext_vector_type(16))) unsigned short v16u;

typedef __attribute__((address_space(1))) v4i* gv4i_p;   // global int4*
typedef __attribute__((address_space(3))) v4i* lv4i_p;   // LDS int4*

#if __has_builtin(__builtin_amdgcn_global_load_async_to_lds_b128)
#define HAVE_ASYNC_LDS 1
#else
#define HAVE_ASYNC_LDS 0
#endif

// Native bf16 convert: let the backend pick v_cvt_pk_bf16_f32 (RNE) if present.
__device__ __forceinline__ __bf16 f2bf16(float f) { return (__bf16)f; }
__device__ __forceinline__ unsigned short f2bfu(float f) {
  return __builtin_bit_cast(unsigned short, (__bf16)f);
}

// B-fragment: two runs of 8 bf16 (k-major) at p and p+16 per lane -> two ds_load_b128
__device__ __forceinline__ v16bf load_bfrag(const unsigned short* p) {
  v8u lo = *(const v8u*)(p);
  v8u hi = *(const v8u*)(p + 16);
  v16u u = __builtin_shufflevector(lo, hi, 0,1,2,3,4,5,6,7,8,9,10,11,12,13,14,15);
  return __builtin_bit_cast(v16bf, u);
}

// LDS strides (elements): 16B-aligned rows, bank-conflict-free N-tile access.
constexpr int W1T_LD = 264;  // bf16 halves per row of W1^T [64][256]
constexpr int W2T_LD = 72;   // bf16 halves per row of W2^T [256][64]
constexpr int XS_LD  = 260;  // floats per row of x tile [16][256]
constexpr int HS_LD  = 72;   // halves per row of h tile [16][64]

__global__ __launch_bounds__(128)
void mlp_mvn_sample(const float* __restrict__ x,  const float* __restrict__ eps,
                    const float* __restrict__ W1, const float* __restrict__ b1,
                    const float* __restrict__ W2, const float* __restrict__ b2,
                    float* __restrict__ out, int nTiles)
{
  __shared__ unsigned short w1t[HDIM * W1T_LD];   // W1^T, bf16: [n][k]
  __shared__ unsigned short w2t[SDIM * W2T_LD];   // W2^T, bf16: [n][k]
  __shared__ float b1s[HDIM];
  __shared__ float b2s[SDIM];
  __shared__ float xs[4][16 * XS_LD];             // per-wave fp32 x tile
  __shared__ unsigned short hs[4][16 * HS_LD];    // per-wave bf16 h tile

  const int tid = threadIdx.x;

  // ---- one-time (per block): weights -> bf16, transposed, into LDS ----
  for (int i = tid; i < SDIM * HDIM; i += 128) {          // W1 [k=256][n=64]
    int k = i >> 6, n = i & 63;
    w1t[n * W1T_LD + k] = f2bfu(W1[i]);
  }
  for (int i = tid; i < HDIM * SDIM; i += 128) {          // W2 [k=64][n=256]
    int k = i >> 8, n = i & 255;
    w2t[n * W2T_LD + k] = f2bfu(W2[i]);
  }
  if (tid < HDIM) b1s[tid] = b1[tid];
  for (int i = tid; i < SDIM; i += 128) b2s[i] = b2[i];
  __syncthreads();

  const int wave  = tid >> 5;
  const int lane  = tid & 31;
  const int lrow  = lane & 15;   // M for A-frags, N for B/C/D frags
  const int khalf = lane >> 4;   // lane-half selects K/M sub-range per ISA layout

  float*          xw = xs[wave];
  unsigned short* hw = hs[wave];

  const v8f vzero = {0.f,0.f,0.f,0.f,0.f,0.f,0.f,0.f};

  const int gwave = blockIdx.x * 4 + wave;
  const int gstep = gridDim.x * 4;

  for (int t = gwave; t < nTiles; t += gstep) {
    const size_t base = (size_t)t * 16 * SDIM;
    const float* xg = x + base;

    // ---- stage 16x256 fp32 x tile (coalesced 512B/instr) ----
#if HAVE_ASYNC_LDS
    // CDNA5 async global->LDS DMA path (ASYNCcnt-tracked, no VGPR round trip)
    #pragma unroll 4
    for (int i = lane; i < 1024; i += 32) {
      __builtin_amdgcn_global_load_async_to_lds_b128(
          (gv4i_p)(xg + i * 4),
          (lv4i_p)(xw + (i >> 6) * XS_LD + (i & 63) * 4),
          0, 0);
    }
#if __has_builtin(__builtin_amdgcn_s_wait_asynccnt)
    __builtin_amdgcn_s_wait_asynccnt(0);
    __asm__ volatile("" ::: "memory");
#else
    __asm__ volatile("s_wait_asynccnt 0x0" ::: "memory");
#endif
#else
    #pragma unroll 4
    for (int i = lane; i < 1024; i += 32) {
      *(v4f*)(xw + (i >> 6) * XS_LD + (i & 63) * 4) = *(const v4f*)(xg + i * 4);
    }
    __asm__ volatile("s_wait_dscnt 0x0" ::: "memory");  // wave-local LDS visibility
#endif

    // ---- GEMM1: [16x256]x[256x64], 8 K-chunks x 4 N-tiles of WMMA ----
    v8f acc1[4] = {vzero, vzero, vzero, vzero};
    #pragma unroll
    for (int kc = 0; kc < 8; ++kc) {
      // A-frag (16-bit layout): lane m=lrow; j<8 -> k=kc*32+khalf*8+j,
      //                          j>=8 -> k=kc*32+16+khalf*8+(j-8)
      const float* ap = xw + lrow * XS_LD + kc * 32 + khalf * 8;
      v4f f0 = *(const v4f*)(ap);
      v4f f1 = *(const v4f*)(ap + 4);
      v4f f2 = *(const v4f*)(ap + 16);
      v4f f3 = *(const v4f*)(ap + 20);
      v16bf A;
      #pragma unroll
      for (int j = 0; j < 4; ++j) {
        A[j]      = f2bf16(f0[j]);
        A[4 + j]  = f2bf16(f1[j]);
        A[8 + j]  = f2bf16(f2[j]);
        A[12 + j] = f2bf16(f3[j]);
      }
      #pragma unroll
      for (int nt = 0; nt < 4; ++nt) {
        v16bf Bf = load_bfrag(w1t + (nt*16 + lrow) * W1T_LD + kc*32 + khalf*8);
        acc1[nt] = __builtin_amdgcn_wmma_f32_16x16x32_bf16(
            false, A, false, Bf, (short)0, acc1[nt], false, false);
      }
    }

    // ---- bias + relu, spill h (bf16) to wave-private LDS for relayout ----
    #pragma unroll
    for (int nt = 0; nt < 4; ++nt) {
      float bias = b1s[nt*16 + lrow];
      #pragma unroll
      for (int r = 0; r < 8; ++r) {          // D-layout: m = khalf*8+r, n = nt*16+lrow
        float hv = fmaxf(acc1[nt][r] + bias, 0.f);
        hw[(khalf*8 + r) * HS_LD + nt*16 + lrow] = f2bfu(hv);
      }
    }
    __asm__ volatile("s_wait_dscnt 0x0" ::: "memory");

    // ---- A-frags for GEMM2 (K=64 -> 2 chunks), from LDS h tile ----
    v16bf A2[2];
    #pragma unroll
    for (int kc = 0; kc < 2; ++kc) {
      const unsigned short* ap2 = hw + lrow * HS_LD + kc*32 + khalf*8;
      v8u lo = *(const v8u*)(ap2);        // k = kc*32 + khalf*8 + [0..7]
      v8u hi = *(const v8u*)(ap2 + 16);   // k = kc*32 + 16 + khalf*8 + [0..7]
      v16u u = __builtin_shufflevector(lo, hi, 0,1,2,3,4,5,6,7,8,9,10,11,12,13,14,15);
      A2[kc] = __builtin_bit_cast(v16bf, u);
    }

    // ---- GEMM2 + fused epilogue: out = x + exp(0.5*logit) * eps ----
    const float* eg = eps + base;
    float*       og = out + base;
    #pragma unroll
    for (int nt = 0; nt < 16; ++nt) {
      v8f acc2 = vzero;
      #pragma unroll
      for (int kc = 0; kc < 2; ++kc) {
        v16bf Bf = load_bfrag(w2t + (nt*16 + lrow) * W2T_LD + kc*32 + khalf*8);
        acc2 = __builtin_amdgcn_wmma_f32_16x16x32_bf16(
            false, A2[kc], false, Bf, (short)0, acc2, false, false);
      }
      const int   n     = nt*16 + lrow;
      const float bias2 = b2s[n];
      #pragma unroll
      for (int r = 0; r < 8; ++r) {
        const int m = khalf*8 + r;
        // sqrt(exp(l)) == exp(0.5*l): one hardware v_exp, no sqrt needed
        float sd = __expf(0.5f * (acc2[r] + bias2));
        og[m * SDIM + n] = xw[m * XS_LD + n] + sd * eg[m * SDIM + n];
      }
    }
  }
}

extern "C" void kernel_launch(void* const* d_in, const int* in_sizes, int n_in,
                              void* d_out, int out_size, void* d_ws, size_t ws_size,
                              hipStream_t stream) {
  const float* x   = (const float*)d_in[0];
  const float* eps = (const float*)d_in[1];
  const float* W1  = (const float*)d_in[2];
  const float* b1  = (const float*)d_in[3];
  const float* W2  = (const float*)d_in[4];
  const float* b2  = (const float*)d_in[5];
  float* out = (float*)d_out;

  const int Bn     = in_sizes[0] / SDIM;   // 131072
  const int nTiles = Bn / 16;              // 8192 row-tiles

  // 512 blocks x 4 waves = 2048 waves -> 4 tiles/wave (grid-stride, safe for any B)
  dim3 grid(512), block(128);
  mlp_mvn_sample<<<grid, block, 0, stream>>>(x, eps, W1, b1, W2, b2, out, nTiles);
}
